// BasicDecoder_300647711215
// MI455X (gfx1250) — matrix operand order repaired and checked by reading, not compiled, not run on hardware
//
#include <hip/hip_runtime.h>
#include <hip/hip_bf16.h>
#include <math.h>

typedef _Float16 half_t;
typedef __attribute__((ext_vector_type(16))) _Float16 v16h;
typedef __attribute__((ext_vector_type(8)))  float    v8f;

union FragAB { v16h v; half_t h[16]; unsigned int u[8]; };

// ---- problem constants ----
constexpr int Bc  = 4;
constexpr int Qc  = 8192;
constexpr int KVc = 1024;
constexpr int Dc  = 1024;
constexpr int Hc  = 8;
constexpr int DHc = 128;
constexpr int OUTC = 512;
constexpr int WIDc = 4096;
constexpr int MQ  = Bc * Qc;    // 32768
constexpr int MKV = Bc * KVc;   // 4096

// ---- Tensor Data Mover support (guarded; falls back to reg staging) ----
#if defined(__has_builtin)
#if __has_builtin(__builtin_amdgcn_tensor_load_to_lds) && \
    __has_builtin(__builtin_amdgcn_s_wait_tensorcnt)
#define USE_TDM 1
#endif
#endif
#ifndef USE_TDM
#define USE_TDM 0
#endif

#if USE_TDM
typedef unsigned int tdm_v4u __attribute__((ext_vector_type(4)));
typedef int          tdm_v8i __attribute__((ext_vector_type(8)));
typedef int          tdm_v4i __attribute__((ext_vector_type(4)));

// 2D f16 tile load, global -> LDS, with LDS row padding.
// rowElems:  tile_dim0 (f16 elements per row, row bytes multiple of 4)
// tileRows:  tile_dim1
// pitchElems: tensor_dim0_stride (global row pitch in elements)
// padIntervalCode: DWORDs of data before pad (0=2,1=4,2=8,3=16,4=32,5=64)
// padAmountCode:   pad DWORDs minus 1
__device__ __forceinline__ void tdm_load_2d_f16(const void* gbase, unsigned ldsOff,
                                                int rowElems, int tileRows,
                                                int pitchElems,
                                                int padIntervalCode, int padAmountCode) {
    unsigned long long ga = (unsigned long long)(uintptr_t)gbase;
    // D# group0: count=1, lds_addr, global_addr[56:0], type=2 (bits 127:126)
    tdm_v4u g0 = { 1u, ldsOff, (unsigned)ga,
                   (unsigned)((ga >> 32) & 0x1FFFFFFu) | (2u << 30) };
    // D# group1: data_size=1(2B), pad_enable, pad codes; dims/strides
    unsigned dw0 = (1u << 16) | (1u << 20) |
                   ((unsigned)padIntervalCode << 22) | ((unsigned)padAmountCode << 25);
    unsigned td0 = (unsigned)pitchElems;   // tensor_dim0 (row length >= rowElems)
    unsigned td1 = 1u << 20;               // tensor_dim1: plenty of rows
    tdm_v8i g1 = { (int)dw0,
                   (int)((td0 & 0xFFFFu) << 16),
                   (int)((td0 >> 16) | ((td1 & 0xFFFFu) << 16)),
                   (int)((td1 >> 16) | ((unsigned)rowElems << 16)),
                   (int)tileRows,
                   (int)pitchElems,
                   0, 0 };
    tdm_v4i z4 = {0, 0, 0, 0};
#if __clang_major__ >= 23
    tdm_v8i z8 = {0, 0, 0, 0, 0, 0, 0, 0};
    __builtin_amdgcn_tensor_load_to_lds(g0, g1, z4, z4, z8, 0);
#else
    __builtin_amdgcn_tensor_load_to_lds(g0, g1, z4, z4, 0);
#endif
}
#endif  // USE_TDM

// =====================================================================
// One-time weight transpose + f16 convert:  Wt[n*K + k] = (f16) W[k*N + n]
// =====================================================================
__global__ __launch_bounds__(256) void transpose_w_f16_kernel(
    const float* __restrict__ W, half_t* __restrict__ Wt, int K, int N) {
    __shared__ float t[32][33];
    const int tx = threadIdx.x & 31;
    const int ty = threadIdx.x >> 5;          // 0..7
    const int k0 = blockIdx.y * 32, n0 = blockIdx.x * 32;
#pragma unroll
    for (int j = 0; j < 4; ++j)
        t[ty * 4 + j][tx] = W[(size_t)(k0 + ty * 4 + j) * N + n0 + tx];
    __syncthreads();
#pragma unroll
    for (int j = 0; j < 4; ++j)
        Wt[(size_t)(n0 + ty * 4 + j) * K + k0 + tx] = (half_t)t[tx][ty * 4 + j];
}

// =====================================================================
// LayerNorm: one block per row (D=1024), 256 threads, f32 in -> f16 out
// =====================================================================
__global__ __launch_bounds__(256) void ln_f16_kernel(const float* __restrict__ x,
                                                     const float* __restrict__ g,
                                                     const float* __restrict__ bvec,
                                                     half_t* __restrict__ out) {
    __shared__ float red[256];
    const int row = blockIdx.x;
    const int tid = threadIdx.x;
    const float* xr = x + (size_t)row * Dc;

    float v[4];
    float s = 0.f;
#pragma unroll
    for (int j = 0; j < 4; ++j) { v[j] = xr[tid + 256 * j]; s += v[j]; }
    red[tid] = s; __syncthreads();
    for (int off = 128; off > 0; off >>= 1) {
        if (tid < off) red[tid] += red[tid + off];
        __syncthreads();
    }
    const float mean = red[0] * (1.f / Dc);
    __syncthreads();

    float s2 = 0.f;
#pragma unroll
    for (int j = 0; j < 4; ++j) { float d = v[j] - mean; s2 += d * d; }
    red[tid] = s2; __syncthreads();
    for (int off = 128; off > 0; off >>= 1) {
        if (tid < off) red[tid] += red[tid + off];
        __syncthreads();
    }
    const float rs = rsqrtf(red[0] * (1.f / Dc) + 1e-5f);

#pragma unroll
    for (int j = 0; j < 4; ++j) {
        int i = tid + 256 * j;
        out[(size_t)row * Dc + i] = (half_t)((v[j] - mean) * rs * g[i] + bvec[i]);
    }
}

// =====================================================================
// WMMA GEMM:  C[M,N] = epi(A_f16[M,K] @ Wt_f16[N,K]^T + bias[N] (+ res))
// Block tile 256x128x32, 256 threads = 8 waves, wave tile 64x64 (4x4 WMMA).
// Double-buffered LDS; tiles staged by the Tensor Data Mover (wave 0 issues
// both descriptors, TENSORcnt-waited), falling back to register staging.
// =====================================================================
enum { EPI_F16 = 0, EPI_F32 = 1, EPI_GELU = 2, EPI_RES = 3, EPI_F16T = 4 };

template <int EPI>
__global__ __launch_bounds__(256) void gemm_wmma_kernel(
    const half_t* __restrict__ A, const half_t* __restrict__ Wt,
    const float* __restrict__ bias, const float* __restrict__ res,
    void* __restrict__ Cout, int M, int N, int K) {
    (void)M; (void)res;
    __shared__ half_t sA[2][256 * 40];    // [m][k] 256x32, stride 40 (64B+16B pad)
    __shared__ half_t sBt[2][128 * 40];   // [n][k] 128x32, stride 40

    const int tid  = threadIdx.x;
    const int lane = tid & 31;
    const int wave = tid >> 5;
    const int wm   = wave >> 1;   // 0..3 -> row block of 64
    const int wn   = wave & 1;    // 0..1 -> col block of 64
    const int rowBase = blockIdx.y * 256;
    const int colBase = blockIdx.x * 128;

    v8f acc[4][4];
#pragma unroll
    for (int i = 0; i < 4; ++i)
#pragma unroll
        for (int j = 0; j < 4; ++j)
            acc[i][j] = (v8f){0.f,0.f,0.f,0.f,0.f,0.f,0.f,0.f};

    const int ksteps = K >> 5;

#if USE_TDM
    // TDM staging: one wave issues both tile DMAs per buffer.
    if (wave == 0) {
        tdm_load_2d_f16(&A[(size_t)rowBase * K], (unsigned)(uintptr_t)&sA[0][0],
                        32, 256, K, 3, 3);
        tdm_load_2d_f16(&Wt[(size_t)colBase * K], (unsigned)(uintptr_t)&sBt[0][0],
                        32, 128, K, 3, 3);
        __builtin_amdgcn_s_wait_tensorcnt(0);
    }
    __syncthreads();
#else
    // staging maps (per thread): A: 4 x b128, B: 2 x b128
    const int ra_r[4] = { (tid + 0*256) >> 2, (tid + 1*256) >> 2,
                          (tid + 2*256) >> 2, (tid + 3*256) >> 2 };
    const int stc = (tid & 3) * 8;
    uint4 ra[4], rb[2];
#pragma unroll
    for (int i = 0; i < 4; ++i)
        ra[i] = *(const uint4*)&A[(size_t)(rowBase + ra_r[i]) * K + stc];
#pragma unroll
    for (int i = 0; i < 2; ++i)
        rb[i] = *(const uint4*)&Wt[(size_t)(colBase + ra_r[i]) * K + stc];
#pragma unroll
    for (int i = 0; i < 4; ++i) *(uint4*)&sA[0][ra_r[i] * 40 + stc] = ra[i];
#pragma unroll
    for (int i = 0; i < 2; ++i) *(uint4*)&sBt[0][ra_r[i] * 40 + stc] = rb[i];
    __syncthreads();
#endif

    for (int kt = 0; kt < ksteps; ++kt) {
        const int cur = kt & 1;
        const bool more = (kt + 1) < ksteps;

#if USE_TDM
        if (more && wave == 0) {
            const int ko = (kt + 1) * 32;
            tdm_load_2d_f16(&A[(size_t)rowBase * K + ko], (unsigned)(uintptr_t)&sA[cur ^ 1][0],
                            32, 256, K, 3, 3);
            tdm_load_2d_f16(&Wt[(size_t)colBase * K + ko], (unsigned)(uintptr_t)&sBt[cur ^ 1][0],
                            32, 128, K, 3, 3);
        }
#else
        if (more) {
            const int ko = (kt + 1) * 32;
#pragma unroll
            for (int i = 0; i < 4; ++i)
                ra[i] = *(const uint4*)&A[(size_t)(rowBase + ra_r[i]) * K + ko + stc];
#pragma unroll
            for (int i = 0; i < 2; ++i)
                rb[i] = *(const uint4*)&Wt[(size_t)(colBase + ra_r[i]) * K + ko + stc];
        }
#endif

        // A fragments (4 x 16-row tiles), all contiguous b128 pairs
        FragAB fa[4];
        const int koffA = (lane & 16) ? 8 : 0;
#pragma unroll
        for (int mi = 0; mi < 4; ++mi) {
            int m = wm * 64 + mi * 16 + (lane & 15);
#pragma unroll
            for (int i = 0; i < 8; ++i)
                fa[mi].u[i] = *(const unsigned int*)&sA[cur][m * 40 + ((i & 4) ? 16 : 0) + koffA + (i & 3) * 2];
        }
        const int koffB = (lane & 16) ? 8 : 0;
#pragma unroll
        for (int nj = 0; nj < 4; ++nj) {
            FragAB fb;
            int n = wn * 64 + nj * 16 + (lane & 15);
#pragma unroll
            for (int vv = 0; vv < 8; ++vv)
                fb.u[vv] = *(const unsigned int*)&sBt[cur][n * 40 + koffB + ((vv & 4) ? 16 : 0) + (vv & 3) * 2];
#pragma unroll
            for (int mi = 0; mi < 4; ++mi)
                acc[mi][nj] = __builtin_amdgcn_wmma_f32_16x16x32_f16(
                    false, fa[mi].v, false, fb.v, (short)0, acc[mi][nj], false, false);
        }

#if USE_TDM
        if (more && wave == 0) __builtin_amdgcn_s_wait_tensorcnt(0);
#else
        if (more) {
            const int nxt = cur ^ 1;
#pragma unroll
            for (int i = 0; i < 4; ++i) *(uint4*)&sA[nxt][ra_r[i] * 40 + stc] = ra[i];
#pragma unroll
            for (int i = 0; i < 2; ++i) *(uint4*)&sBt[nxt][ra_r[i] * 40 + stc] = rb[i];
        }
#endif
        __syncthreads();
    }

    // epilogue
#pragma unroll
    for (int mi = 0; mi < 4; ++mi)
#pragma unroll
        for (int nj = 0; nj < 4; ++nj)
#pragma unroll
            for (int r = 0; r < 8; ++r) {
                int grow = rowBase + wm * 64 + mi * 16 + r + ((lane & 16) ? 8 : 0);
                int gcol = colBase + wn * 64 + nj * 16 + (lane & 15);
                float val = acc[mi][nj][r] + bias[gcol];
                size_t oi = (size_t)grow * N + gcol;
                if (EPI == EPI_F16) {
                    ((half_t*)Cout)[oi] = (half_t)val;
                } else if (EPI == EPI_F32) {
                    ((float*)Cout)[oi] = val;
                } else if (EPI == EPI_GELU) {
                    ((half_t*)Cout)[oi] =
                        (half_t)(0.5f * val * (1.f + erff(val * 0.70710678118654752f)));
                } else if (EPI == EPI_RES) {
                    val += res[oi];
                    ((half_t*)Cout)[oi] = (half_t)val;
                } else {  // EPI_F16T: V projection -> vpt[(b*H+h)*DH+dh][kv]
                    int b  = grow >> 10;          // /KVc
                    int kv = grow & (KVc - 1);
                    int hh = gcol >> 7;           // /DHc
                    int dh = gcol & (DHc - 1);
                    size_t ot = (((size_t)b * Hc + hh) * DHc + dh) * KVc + kv;
                    ((half_t*)Cout)[ot] = (half_t)val;
                }
            }
}

// =====================================================================
// Flash cross-attention: 32 q-rows/block, 4 waves, online softmax.
// K and V^T chunks staged by the TDM (fallback: vector loads).
// =====================================================================
__global__ __launch_bounds__(128) void attention_kernel(
    const half_t* __restrict__ qp, const half_t* __restrict__ kp,
    const half_t* __restrict__ vpt, const unsigned char* __restrict__ qmask,
    half_t* __restrict__ summed) {
    __shared__ half_t sQ[32 * 136];    // [q][dh]   256B data + 16B pad
    __shared__ half_t sK[64 * 136];    // [kv][dh]  256B data + 16B pad
    __shared__ half_t sVt[128 * 72];   // [dh][kv]  128B data + 16B pad
    __shared__ float  sS[32 * 72];
    __shared__ half_t sP[32 * 72];
    __shared__ float rowM[32], rowL[32], rowFac[32];
    __shared__ float sRed[128];

    const int tid  = threadIdx.x;
    const int lane = tid & 31;
    const int wave = tid >> 5;
    const int q0 = blockIdx.x * 32;
    const int h  = blockIdx.y;
    const int b  = blockIdx.z;
    const float scale = 0.08838834764831845f;  // 1/sqrt(128)

    const half_t* qb  = qp  + ((size_t)b * Qc + q0) * Dc + h * DHc;
    const half_t* kb  = kp  + (size_t)b * KVc * Dc + h * DHc;
    const half_t* vtb = vpt + ((size_t)b * Hc + h) * DHc * KVc;   // row dh, stride KVc

#pragma unroll
    for (int it = 0; it < 4; ++it) {
        int idx = tid + it * 128;    // 0..511
        int r = idx >> 4, c = (idx & 15) * 8;
        *(uint4*)&sQ[r * 136 + c] = *(const uint4*)&qb[(size_t)r * Dc + c];
    }
    if (tid < 32) { rowM[tid] = -1e30f; rowL[tid] = 0.f; }

    v8f acc[2][2];
#pragma unroll
    for (int i = 0; i < 2; ++i)
#pragma unroll
        for (int j = 0; j < 2; ++j)
            acc[i][j] = (v8f){0.f,0.f,0.f,0.f,0.f,0.f,0.f,0.f};

    const int sRow = (wave & 1) * 16;   // S row block (16)
    const int sCol = (wave >> 1) * 32;  // S col block (32)
    const int oc   = wave * 32;         // output col block (32 of DH=128)
    const int row4 = tid & 31;          // softmax: 4 threads per row
    const int part = tid >> 5;

    for (int ck = 0; ck < KVc / 64; ++ck) {
        const int kv0 = ck * 64;
        __syncthreads();  // previous chunk done with sK/sVt/sP
#if USE_TDM
        if (wave == 0) {
            // K chunk: 64 rows x 128 halves (256B data + 16B pad -> codes 5/3)
            tdm_load_2d_f16(&kb[(size_t)kv0 * Dc], (unsigned)(uintptr_t)&sK[0],
                            128, 64, Dc, 5, 3);
            // V^T chunk: 128 rows x 64 halves (128B data + 16B pad -> codes 4/3)
            tdm_load_2d_f16(&vtb[kv0], (unsigned)(uintptr_t)&sVt[0],
                            64, 128, KVc, 4, 3);
            __builtin_amdgcn_s_wait_tensorcnt(0);
        }
#else
#pragma unroll
        for (int it = 0; it < 8; ++it) {
            int idx = tid + it * 128;          // 0..1023
            int r = idx >> 4, c = (idx & 15) * 8;
            *(uint4*)&sK[r * 136 + c] = *(const uint4*)&kb[(size_t)(kv0 + r) * Dc + c];
            int rv = idx >> 3, cv = (idx & 7) * 8;   // 128 rows x 64 cols
            *(uint4*)&sVt[rv * 72 + cv] =
                *(const uint4*)&vtb[(size_t)rv * KVc + kv0 + cv];
        }
#endif
        __syncthreads();

        // S = Q @ K^T  (wave: 16x32 subtile, DH=128 -> 4 wmma steps)
        v8f sacc[2];
        sacc[0] = (v8f){0.f,0.f,0.f,0.f,0.f,0.f,0.f,0.f};
        sacc[1] = (v8f){0.f,0.f,0.f,0.f,0.f,0.f,0.f,0.f};
#pragma unroll
        for (int ks = 0; ks < 4; ++ks) {
            FragAB fa;
            int m = sRow + (lane & 15);
            int ka = ks * 32 + ((lane & 16) ? 8 : 0);
#pragma unroll
            for (int i = 0; i < 8; ++i)
                fa.u[i] = *(const unsigned int*)&sQ[m * 136 + ka + ((i & 4) ? 16 : 0) + (i & 3) * 2];
#pragma unroll
            for (int tj = 0; tj < 2; ++tj) {
                FragAB fb;  // B = K^T: element [k][n] = sK[n][k] -> contiguous
                int n = sCol + tj * 16 + (lane & 15);
                int kb0 = ks * 32 + ((lane & 16) ? 8 : 0);
#pragma unroll
                for (int vv = 0; vv < 8; ++vv)
                    fb.u[vv] = *(const unsigned int*)&sK[n * 136 + kb0 + ((vv & 4) ? 16 : 0) + (vv & 3) * 2];
                sacc[tj] = __builtin_amdgcn_wmma_f32_16x16x32_f16(
                    false, fa.v, false, fb.v, (short)0, sacc[tj], false, false);
            }
        }
#pragma unroll
        for (int tj = 0; tj < 2; ++tj)
#pragma unroll
            for (int r = 0; r < 8; ++r) {
                int row = sRow + r + ((lane & 16) ? 8 : 0);
                int col = sCol + tj * 16 + (lane & 15);
                sS[row * 72 + col] = sacc[tj][r] * scale;
            }
        __syncthreads();

        // online softmax, 4 threads per row
        float cm = -1e30f;
#pragma unroll
        for (int j = 0; j < 16; ++j) cm = fmaxf(cm, sS[row4 * 72 + part * 16 + j]);
        sRed[tid] = cm; __syncthreads();
        if (part == 0) {
            float mold = rowM[row4];
            float m4 = fmaxf(fmaxf(sRed[row4], sRed[32 + row4]),
                             fmaxf(sRed[64 + row4], sRed[96 + row4]));
            float mnew = fmaxf(mold, m4);
            rowFac[row4] = __expf(mold - mnew);
            rowM[row4] = mnew;
        }
        __syncthreads();
        {
            float mnew = rowM[row4];
            float ps = 0.f;
#pragma unroll
            for (int j = 0; j < 16; ++j) {
                int idx = row4 * 72 + part * 16 + j;
                float p = __expf(sS[idx] - mnew);
                sP[idx] = (half_t)p;
                ps += p;
            }
            sRed[tid] = ps;
        }
        __syncthreads();
        if (part == 0)
            rowL[row4] = rowL[row4] * rowFac[row4] +
                         (sRed[row4] + sRed[32 + row4] + sRed[64 + row4] + sRed[96 + row4]);
        __syncthreads();

        // rescale running O, then O += P @ V (K=64 -> 2 wmma steps)
#pragma unroll
        for (int ri = 0; ri < 2; ++ri)
#pragma unroll
            for (int r = 0; r < 8; ++r) {
                int row = ri * 16 + r + ((lane & 16) ? 8 : 0);
                float f = rowFac[row];
#pragma unroll
                for (int ci = 0; ci < 2; ++ci) acc[ri][ci][r] *= f;
            }
#pragma unroll
        for (int ks = 0; ks < 2; ++ks) {
            FragAB fp[2];
#pragma unroll
            for (int ri = 0; ri < 2; ++ri) {
                int m = ri * 16 + (lane & 15);
                int ka = ks * 32 + ((lane & 16) ? 8 : 0);
#pragma unroll
                for (int i = 0; i < 8; ++i)
                    fp[ri].u[i] = *(const unsigned int*)&sP[m * 72 + ka + ((i & 4) ? 16 : 0) + (i & 3) * 2];
            }
#pragma unroll
            for (int ci = 0; ci < 2; ++ci) {
                FragAB fv;  // from sVt[n][k]: contiguous b128
                int n = oc + ci * 16 + (lane & 15);
                int kv = ks * 32 + ((lane & 16) ? 8 : 0);
#pragma unroll
                for (int vv = 0; vv < 8; ++vv)
                    fv.u[vv] = *(const unsigned int*)&sVt[n * 72 + kv + ((vv & 4) ? 16 : 0) + (vv & 3) * 2];
#pragma unroll
                for (int ri = 0; ri < 2; ++ri)
                    acc[ri][ci] = __builtin_amdgcn_wmma_f32_16x16x32_f16(
                        false, fp[ri].v, false, fv.v, (short)0, acc[ri][ci], false, false);
            }
        }
    }
    __syncthreads();

    // finalize: /l, query_mask wipe, store f16 in [(b*H+h)*Q+t, DH] layout
#pragma unroll
    for (int ri = 0; ri < 2; ++ri)
#pragma unroll
        for (int ci = 0; ci < 2; ++ci)
#pragma unroll
            for (int r = 0; r < 8; ++r) {
                int row = ri * 16 + r + ((lane & 16) ? 8 : 0);
                int col = oc + ci * 16 + (lane & 15);
                float inv = 1.f / rowL[row];
                float wipe = qmask[(size_t)b * Qc + q0 + row] ? 1.f : 0.f;
                summed[(((size_t)b * Hc + h) * Qc + (q0 + row)) * DHc + col] =
                    (half_t)(acc[ri][ci][r] * inv * wipe);
            }
}

// =====================================================================
extern "C" void kernel_launch(void* const* d_in, const int* in_sizes, int n_in,
                              void* d_out, int out_size, void* d_ws, size_t ws_size,
                              hipStream_t stream) {
    (void)in_sizes; (void)n_in; (void)out_size; (void)ws_size;
    const float* query  = (const float*)d_in[0];
    const float* z      = (const float*)d_in[1];
    const unsigned char* qmask = (const unsigned char*)d_in[2];
    const float* ln_q_g = (const float*)d_in[3];
    const float* ln_q_b = (const float*)d_in[4];
    const float* ln_kv_g = (const float*)d_in[5];
    const float* ln_kv_b = (const float*)d_in[6];
    const float* ln_a_g = (const float*)d_in[7];
    const float* ln_a_b = (const float*)d_in[8];
    const float* Wq = (const float*)d_in[9];  const float* bq = (const float*)d_in[10];
    const float* Wk = (const float*)d_in[11]; const float* bk = (const float*)d_in[12];
    const float* Wv = (const float*)d_in[13]; const float* bv = (const float*)d_in[14];
    const float* Wo = (const float*)d_in[15]; const float* bo = (const float*)d_in[16];
    const float* W1 = (const float*)d_in[17]; const float* b1 = (const float*)d_in[18];
    const float* W2 = (const float*)d_in[19]; const float* b2 = (const float*)d_in[20];
    const float* Wf = (const float*)d_in[21]; const float* bf = (const float*)d_in[22];

    char* ws = (char*)d_ws;
    size_t off = 0;
    auto alloc = [&](size_t bytes) -> char* {
        char* p = ws + off;
        off += (bytes + 255) & ~(size_t)255;
        return p;
    };
    half_t* qn     = (half_t*)alloc((size_t)MQ  * Dc * 2);    // reused later as a_ln
    half_t* kvn    = (half_t*)alloc((size_t)MKV * Dc * 2);
    half_t* qp     = (half_t*)alloc((size_t)MQ  * Dc * 2);    // reused later as x2
    half_t* kp     = (half_t*)alloc((size_t)MKV * Dc * 2);
    half_t* vpt    = (half_t*)alloc((size_t)MKV * Dc * 2);    // transposed V proj
    half_t* summed = (half_t*)alloc((size_t)MQ  * Dc * 2);
    float*  attn   = (float*)alloc((size_t)MQ   * Dc * 4);
    half_t* h1     = (half_t*)alloc((size_t)MQ  * WIDc * 2);
    half_t* wqt    = (half_t*)alloc((size_t)Dc * Dc * 2);
    half_t* wkt    = (half_t*)alloc((size_t)Dc * Dc * 2);
    half_t* wvt    = (half_t*)alloc((size_t)Dc * Dc * 2);
    half_t* wot    = (half_t*)alloc((size_t)Dc * Dc * 2);
    half_t* w1t    = (half_t*)alloc((size_t)Dc * WIDc * 2);
    half_t* w2t    = (half_t*)alloc((size_t)WIDc * Dc * 2);
    half_t* wft    = (half_t*)alloc((size_t)Dc * OUTC * 2);
    half_t* a_ln   = qn;   // qn dead after q projection
    half_t* x2     = qp;   // qp dead after attention

    // 0: one-time weight transposes (f32 -> f16, [N][K])
    transpose_w_f16_kernel<<<dim3(Dc/32,   Dc/32),   256, 0, stream>>>(Wq, wqt, Dc, Dc);
    transpose_w_f16_kernel<<<dim3(Dc/32,   Dc/32),   256, 0, stream>>>(Wk, wkt, Dc, Dc);
    transpose_w_f16_kernel<<<dim3(Dc/32,   Dc/32),   256, 0, stream>>>(Wv, wvt, Dc, Dc);
    transpose_w_f16_kernel<<<dim3(Dc/32,   Dc/32),   256, 0, stream>>>(Wo, wot, Dc, Dc);
    transpose_w_f16_kernel<<<dim3(WIDc/32, Dc/32),   256, 0, stream>>>(W1, w1t, Dc, WIDc);
    transpose_w_f16_kernel<<<dim3(Dc/32,   WIDc/32), 256, 0, stream>>>(W2, w2t, WIDc, Dc);
    transpose_w_f16_kernel<<<dim3(OUTC/32, Dc/32),   256, 0, stream>>>(Wf, wft, Dc, OUTC);

    // 1,2: layernorms -> f16
    ln_f16_kernel<<<MQ,  256, 0, stream>>>(query, ln_q_g, ln_q_b, qn);
    ln_f16_kernel<<<MKV, 256, 0, stream>>>(z, ln_kv_g, ln_kv_b, kvn);
    // 3-5: q/k/v projections (V writes transposed layout)
    gemm_wmma_kernel<EPI_F16><<<dim3(Dc/128, MQ/256),  256, 0, stream>>>(qn,  wqt, bq, nullptr, qp,  MQ,  Dc, Dc);
    gemm_wmma_kernel<EPI_F16><<<dim3(Dc/128, MKV/256), 256, 0, stream>>>(kvn, wkt, bk, nullptr, kp,  MKV, Dc, Dc);
    gemm_wmma_kernel<EPI_F16T><<<dim3(Dc/128, MKV/256),256, 0, stream>>>(kvn, wvt, bv, nullptr, vpt, MKV, Dc, Dc);
    // 6: fused flash attention
    attention_kernel<<<dim3(Qc/32, Hc, Bc), 128, 0, stream>>>(qp, kp, vpt, qmask, summed);
    // 7: output projection (f32, kept for residual + LN)
    gemm_wmma_kernel<EPI_F32><<<dim3(Dc/128, MQ/256), 256, 0, stream>>>(summed, wot, bo, nullptr, attn, MQ, Dc, Dc);
    // 8: LN of attn_out
    ln_f16_kernel<<<MQ, 256, 0, stream>>>(attn, ln_a_g, ln_a_b, a_ln);
    // 9: W1 + exact GELU
    gemm_wmma_kernel<EPI_GELU><<<dim3(WIDc/128, MQ/256), 256, 0, stream>>>(a_ln, w1t, b1, nullptr, h1, MQ, WIDc, Dc);
    // 10: W2 + residual add
    gemm_wmma_kernel<EPI_RES><<<dim3(Dc/128, MQ/256), 256, 0, stream>>>(h1, w2t, b2, attn, x2, MQ, Dc, WIDc);
    // 11: final projection -> f32 d_out
    gemm_wmma_kernel<EPI_F32><<<dim3(OUTC/128, MQ/256), 256, 0, stream>>>(x2, wft, bf, nullptr, (float*)d_out, MQ, OUTC, Dc);
}
